// RVQBottleneck_16312285791125
// MI455X (gfx1250) — matrix operand order repaired and checked by reading, not compiled, not run on hardware
//
#include <hip/hip_runtime.h>
#include <stdint.h>

typedef _Float16 v16h  __attribute__((ext_vector_type(16)));
typedef _Float16 half8 __attribute__((ext_vector_type(8)));
typedef _Float16 half4v __attribute__((ext_vector_type(4)));
typedef float    v8f   __attribute__((ext_vector_type(8)));
typedef int      v4i   __attribute__((ext_vector_type(4)));

#define D           512
#define NCODE       1024
#define TOK_PER_BLK 64
#define NITER       (NCODE / 32)   // 32 iterations, 2 column-tiles each

// gfx1250 async global->LDS copy path (ASYNCcnt), if the toolchain exposes it
#if defined(__has_builtin)
#if __has_builtin(__builtin_amdgcn_global_load_async_to_lds_b128) && \
    __has_builtin(__builtin_amdgcn_s_wait_asynccnt)
#define USE_ASYNC_LDS 1
#endif
#endif
#ifndef USE_ASYNC_LDS
#define USE_ASYNC_LDS 0
#endif

#if USE_ASYNC_LDS
// b128 async copy: per-lane 16B  global -> LDS, tracked by ASYNCcnt.
// Param types per hipcc diagnostic: (v4i AS1*, v4i AS3*, imm offset, imm cpol)
#define ASYNC_CP16(gsrc, ldst)                                                \
  __builtin_amdgcn_global_load_async_to_lds_b128(                             \
      (__attribute__((address_space(1))) v4i*)(uintptr_t)(const void*)(gsrc), \
      (__attribute__((address_space(3))) v4i*)(ldst), 0, 0)
#endif

// ---------------------------------------------------------------------------
// Prep: split codebooks into f16 hi/lo halves (near-fp32 WMMA dot products)
// and compute per-row squared norms in fp32.  grid = 2*NCODE x 32 threads.
// ---------------------------------------------------------------------------
__global__ void __launch_bounds__(32)
rvq_prep(const float* __restrict__ e0, const float* __restrict__ e1,
         _Float16* __restrict__ ebh, _Float16* __restrict__ ebl,
         float* __restrict__ e2) {
  const int b     = blockIdx.x;
  const int stage = b >> 10;
  const int row   = b & (NCODE - 1);
  const float* src = stage ? e1 : e0;
  const int lane  = threadIdx.x;
  const int base  = (stage * NCODE + row) * D;

  float acc = 0.f;
  for (int i = 0; i < D / 32; ++i) {
    int k = lane + 32 * i;
    float v = src[row * D + k];
    acc += v * v;
    _Float16 hi = (_Float16)v;
    ebh[base + k] = hi;
    ebl[base + k] = (_Float16)(v - (float)hi);
  }
  for (int m = 16; m >= 1; m >>= 1) acc += __shfl_xor(acc, m, 32);
  if (lane == 0) e2[stage * NCODE + row] = acc;
}

// ---------------------------------------------------------------------------
// Main RVQ kernel: 128 threads = 4 waves, 64 tokens per block.
// Double-buffered 32-row B tiles staged asynchronously while computing.
// ---------------------------------------------------------------------------
__global__ void __launch_bounds__(128)
rvq_main(const float* __restrict__ x,
         const float* __restrict__ e0f, const float* __restrict__ e1f,
         const _Float16* __restrict__ ebh, const _Float16* __restrict__ ebl,
         const float* __restrict__ e2, float* __restrict__ out) {
  __shared__ _Float16 Xh[TOK_PER_BLK][D];   // 64 KB
  __shared__ _Float16 Xl[TOK_PER_BLK][D];   // 64 KB
  __shared__ _Float16 Bh[2][32][D];         // 64 KB (double buffer)
  __shared__ _Float16 Bl[2][32][D];         // 64 KB   -> 256 KB total (<=320KB/WGP)

  const int tid  = threadIdx.x;
  const int lane = tid & 31;
  const int w    = tid >> 5;       // wave id 0..3
  const int r    = lane & 15;      // row (A) / column (B,C) within 16
  const int h    = lane >> 4;      // half-wave id
  const int m0   = blockIdx.x * TOK_PER_BLK;
  const int arow = 16 * w + r;     // this lane's A row in LDS

  // cooperative stage of 32 pre-split codebook rows into LDS buffer `buf`
  auto stageB = [&](const _Float16* EH, const _Float16* EL, int buf, int n0) {
    for (int i = 0; i < 16; ++i) {
      int c   = tid + 128 * i;      // 2048 chunks of 8 halves per array
      int row = c >> 6;
      int off = (c & 63) * 8;
#if USE_ASYNC_LDS
      ASYNC_CP16(&EH[(size_t)(n0 + row) * D + off], &Bh[buf][row][off]);
      ASYNC_CP16(&EL[(size_t)(n0 + row) * D + off], &Bl[buf][row][off]);
#else
      *(half8*)&Bh[buf][row][off] = *(const half8*)&EH[(size_t)(n0 + row) * D + off];
      *(half8*)&Bl[buf][row][off] = *(const half8*)&EL[(size_t)(n0 + row) * D + off];
#endif
    }
  };

  // ---- stage + split x tile into LDS (coalesced float4) ----
  for (int i = 0; i < 64; ++i) {
    int f   = tid + 128 * i;          // 8192 float4: 64 rows x 128 f4/row
    int row = f >> 7;
    int c4  = (f & 127) * 4;
    const float4 v = *(const float4*)&x[(size_t)(m0 + row) * D + c4];
    float vv[4] = {v.x, v.y, v.z, v.w};
    half4v hh, hl;
    for (int j = 0; j < 4; ++j) {
      _Float16 hi = (_Float16)vv[j];
      hh[j] = hi;
      hl[j] = (_Float16)(vv[j] - (float)hi);
    }
    *(half4v*)&Xh[row][c4] = hh;
    *(half4v*)&Xl[row][c4] = hl;
  }

  int idxs[2][8];

  for (int s = 0; s < 2; ++s) {
    const _Float16* EH = ebh + (size_t)s * NCODE * D;
    const _Float16* EL = ebl + (size_t)s * NCODE * D;
    const float*    E2 = e2 + s * NCODE;

    float best[8];
    int   bidx[8];
    for (int v = 0; v < 8; ++v) { best[v] = -3.0e38f; bidx[v] = 0; }

    stageB(EH, EL, 0, 0);            // prime the pipeline

    for (int it = 0; it < NITER; ++it) {
      const int n0 = it * 32;
      const int bb = it & 1;
#if USE_ASYNC_LDS
      __builtin_amdgcn_s_wait_asynccnt(0);   // our staged tile has landed in LDS
#endif
      __syncthreads();                       // everyone's tile `it` ready; compute it-1 done

      if (it + 1 < NITER) stageB(EH, EL, bb ^ 1, n0 + 32);   // overlap with compute

      // prefetch tile it+2 toward the caches (it+1 already in flight)
      if (it + 2 < NITER) {
        const char* PH = (const char*)&EH[(size_t)(n0 + 64) * D];
        const char* PL = (const char*)&EL[(size_t)(n0 + 64) * D];
        for (int p = 0; p < 2; ++p) {        // 32KB/array, 128B lines
          size_t off = (size_t)(tid + 128 * p) * 128;
          __builtin_prefetch(PH + off, 0, 2);
          __builtin_prefetch(PL + off, 0, 2);
        }
      }

      v8f C0 = {0.f,0.f,0.f,0.f,0.f,0.f,0.f,0.f};
      v8f C1 = {0.f,0.f,0.f,0.f,0.f,0.f,0.f,0.f};
      for (int k0 = 0; k0 < D; k0 += 32) {
        // A fragment (16-bit A 16x32: contiguous K range [k0+16h,+16))
        v16h aH = *(const v16h*)&Xh[arow][k0 + 16 * h];
        v16h aL = *(const v16h*)&Xl[arow][k0 + 16 * h];
        // B fragments for both column tiles (contiguous K range [k0+16h,+16))
        v16h bH0 = *(const v16h*)&Bh[bb][r][k0 + 16 * h];
        v16h bL0 = *(const v16h*)&Bl[bb][r][k0 + 16 * h];
        v16h bH1 = *(const v16h*)&Bh[bb][16 + r][k0 + 16 * h];
        v16h bL1 = *(const v16h*)&Bl[bb][16 + r][k0 + 16 * h];
        // split-precision dot, two independent accumulation chains
        C0 = __builtin_amdgcn_wmma_f32_16x16x32_f16(false, aL, false, bH0, (short)0, C0, false, false);
        C1 = __builtin_amdgcn_wmma_f32_16x16x32_f16(false, aL, false, bH1, (short)0, C1, false, false);
        C0 = __builtin_amdgcn_wmma_f32_16x16x32_f16(false, aH, false, bL0, (short)0, C0, false, false);
        C1 = __builtin_amdgcn_wmma_f32_16x16x32_f16(false, aH, false, bL1, (short)0, C1, false, false);
        C0 = __builtin_amdgcn_wmma_f32_16x16x32_f16(false, aH, false, bH0, (short)0, C0, false, false);
        C1 = __builtin_amdgcn_wmma_f32_16x16x32_f16(false, aH, false, bH1, (short)0, C1, false, false);
      }

      const float e2a = E2[n0 + r];
      const float e2b = E2[n0 + 16 + r];
      for (int v = 0; v < 8; ++v) {
        float sc0 = 2.0f * C0[v] - e2a;        // argmax(2xe - e^2) == argmin dist
        if (sc0 > best[v]) { best[v] = sc0; bidx[v] = n0 + r; }
        float sc1 = 2.0f * C1[v] - e2b;
        if (sc1 > best[v]) { best[v] = sc1; bidx[v] = n0 + 16 + r; }
      }
    }

    // argmax reduction across the 16 lanes of each half-wave (ties -> lowest idx)
    for (int mask = 1; mask < 16; mask <<= 1) {
      for (int v = 0; v < 8; ++v) {
        float os = __shfl_xor(best[v], mask, 32);
        int   oi = __shfl_xor(bidx[v], mask, 32);
        if (os > best[v] || (os == best[v] && oi < bidx[v])) { best[v] = os; bidx[v] = oi; }
      }
    }
    for (int v = 0; v < 8; ++v) idxs[s][v] = bidx[v];

    if (s == 0) {
      // residual = x - emb0[i0]; re-split hi/lo in LDS (wave-local rows)
      for (int v = 0; v < 8; ++v) {
        int lrow = 16 * w + v + 8 * h;
        const float* er = e0f + (size_t)bidx[v] * D;
        for (int t = 0; t < 32; ++t) {
          int k = r + 16 * t;
          float xv = (float)Xh[lrow][k] + (float)Xl[lrow][k] - er[k];
          _Float16 hi = (_Float16)xv;
          Xh[lrow][k] = hi;
          Xl[lrow][k] = (_Float16)(xv - (float)hi);
        }
      }
      __syncthreads();
    }
  }

  // decode in exact fp32: out = emb0[i0] + emb1[i1]
  for (int v = 0; v < 8; ++v) {
    int grow = m0 + 16 * w + v + 8 * h;
    const float* p0 = e0f + (size_t)idxs[0][v] * D;
    const float* p1 = e1f + (size_t)idxs[1][v] * D;
    for (int t = 0; t < 8; ++t) {
      int k4 = (r + 16 * t) * 4;
      float4 a = *(const float4*)&p0[k4];
      float4 b = *(const float4*)&p1[k4];
      float4 o;
      o.x = a.x + b.x; o.y = a.y + b.y; o.z = a.z + b.z; o.w = a.w + b.w;
      *(float4*)&out[(size_t)grow * D + k4] = o;
    }
  }
}

// ---------------------------------------------------------------------------
extern "C" void kernel_launch(void* const* d_in, const int* in_sizes, int n_in,
                              void* d_out, int out_size, void* d_ws, size_t ws_size,
                              hipStream_t stream) {
  (void)n_in; (void)out_size; (void)ws_size;
  const float* x  = (const float*)d_in[0];
  const float* e0 = (const float*)d_in[1];
  const float* e1 = (const float*)d_in[2];
  float* out = (float*)d_out;

  char* ws = (char*)d_ws;
  const size_t nEmb = (size_t)2 * NCODE * D;                    // split-array elems
  _Float16* ebh = (_Float16*)ws;                                // 2 MB
  _Float16* ebl = (_Float16*)(ws + nEmb * sizeof(_Float16));    // 2 MB
  float*    e2v = (float*)(ws + 2 * nEmb * sizeof(_Float16));   // 8 KB

  const int M = in_sizes[0] / D;  // 32768 tokens

  rvq_prep<<<2 * NCODE, 32, 0, stream>>>(e0, e1, ebh, ebl, e2v);
  rvq_main<<<M / TOK_PER_BLK, 128, 0, stream>>>(x, e0, e1, ebh, ebl, e2v, out);
}
